// EdgeConvolution_28192165331141
// MI455X (gfx1250) — matrix operand order repaired and checked by reading, not compiled, not run on hardware
//
#include <hip/hip_runtime.h>

#define N_NODES 50000
#define N_EDGES 1600000
#define N_MACRO (N_EDGES / 32)     // 50000 32-edge macro-tiles (exact)
#define NODE_TILES (N_NODES / 16)  // 3125 node tiles (exact)

typedef __attribute__((ext_vector_type(16))) __bf16    v16bf;
typedef __attribute__((ext_vector_type(8)))  float     v8f;
typedef __attribute__((ext_vector_type(4)))  unsigned  v4u;
typedef __attribute__((ext_vector_type(2)))  float     v2f;

__device__ __forceinline__ unsigned short f2bf(float f) {
  unsigned u = __float_as_uint(f);
  u += 0x7FFFu + ((u >> 16) & 1u);          // round-to-nearest-even
  return (unsigned short)(u >> 16);
}
__device__ __forceinline__ float bf2f(unsigned short h) {
  return __uint_as_float(((unsigned)h) << 16);
}
// swish via v_rcp_f32 (TRANS) instead of IEEE divide chain
__device__ __forceinline__ float swishf(float x) {
  return x * __builtin_amdgcn_rcpf(1.0f + __expf(-x));
}

// A fragment (16-bit A 16x32, ISA 7.12.2): lane row = lane%16; lanes<16 take K
// offsets [0..8)+[16..24), lanes>=16 take [8..16)+[24..32) -> two 16B chunks 32B apart.
__device__ __forceinline__ v16bf load_frag_a(const unsigned short* p) {
  v16bf r;
  ((v4u*)&r)[0] = ((const v4u*)p)[0];
  ((v4u*)&r)[1] = ((const v4u*)p)[2];
  return r;
}
// B fragment: pre-swizzled, each lane's 16 bf16 contiguous (32B).
__device__ __forceinline__ v16bf load_frag_b(const unsigned short* p) {
  v16bf r;
  ((v4u*)&r)[0] = ((const v4u*)p)[0];
  ((v4u*)&r)[1] = ((const v4u*)p)[1];
  return r;
}
__device__ __forceinline__ v8f wmma_bf16(v16bf a, v16bf b, v8f c) {
  return __builtin_amdgcn_wmma_f32_16x16x32_bf16(false, a, false, b, (short)0, c,
                                                 false, false);
}
__device__ __forceinline__ float dec_max(unsigned u) {
  return __uint_as_float((u & 0x80000000u) ? (u & 0x7FFFFFFFu) : ~u);
}

// ---------------------------------------------------------------- prep ------
// Swizzle W[K][64] (row-major f32) into bf16 B-fragment layout.
__device__ __forceinline__ unsigned short swz(const float* src, int d) {
  int h = d & 1, j = (d >> 1) & 7, lane = (d >> 4) & 31, ft = d >> 9;
  int nt = ft & 3, kt = ft >> 2;
  int ksrc = kt * 32 + ((lane < 16) ? 0 : 8) + 2 * j + h + ((j >= 4) ? 8 : 0);
  int n = nt * 16 + (lane & 15);
  return f2bf(src[ksrc * 64 + n]);
}

__global__ __launch_bounds__(256) void prep_kernel(
    const float* W1, const float* b1, const float* W2, const float* b2,
    const float* Wa1, const float* ba1, const float* Wa2, const float* ba2,
    const float* Wu1, const float* bu1, const float* Wu2, const float* bu2,
    unsigned* scal, float* bias,
    unsigned short* W1sw, unsigned short* Wa1sw, unsigned short* W2sw,
    unsigned short* Wu1sw, unsigned short* Wu2sw) {
  int tid = blockIdx.x * 256 + threadIdx.x;
  int stride = gridDim.x * 256;
  if (tid == 0) { scal[0] = 0u; ((float*)scal)[1] = 0.0f; bias[384] = ba2[0]; }
  if (tid < 64) {
    bias[tid] = b1[tid];        bias[64 + tid] = b2[tid];
    bias[128 + tid] = ba1[tid]; bias[192 + tid] = Wa2[tid];
    bias[256 + tid] = bu1[tid]; bias[320 + tid] = bu2[tid];
  }
  for (int i = tid; i < 10240; i += stride) W1sw[i]  = swz(W1, i);   // 160x64
  for (int i = tid; i < 10240; i += stride) Wa1sw[i] = swz(Wa1, i);  // 160x64
  for (int i = tid; i < 4096;  i += stride) W2sw[i]  = swz(W2, i);   // 64x64
  for (int i = tid; i < 8192;  i += stride) Wu1sw[i] = swz(Wu1, i);  // 128x64
  for (int i = tid; i < 4096;  i += stride) Wu2sw[i] = swz(Wu2, i);  // 64x64
}

// ---------------------------------------------------------- edge pass ------
// 128 threads = 4 waves; each wave owns a 32-edge macro-tile (two 16-row A
// tiles) so every B fragment load feeds TWO WMMAs (halves WGP$/L2 B traffic).
__global__ __launch_bounds__(128) void edge_kernel(
    const float* __restrict__ nodeF, const float* __restrict__ edgeF,
    const int* __restrict__ senders, const int* __restrict__ receivers,
    const unsigned short* __restrict__ W1sw,
    const unsigned short* __restrict__ Wa1sw,
    const unsigned short* __restrict__ W2sw, const float* __restrict__ bias,
    float* __restrict__ logits, unsigned short* __restrict__ msgs,
    unsigned* __restrict__ maxbits) {
  __shared__ __align__(16) unsigned short sA[4][32 * 168];  // 32x160 bf16, padded
  __shared__ __align__(16) unsigned short sH[4][32 * 72];   // 32x64 bf16, padded
  __shared__ float sb1[64], sb2[64], sba1[64], swa2[64];
  __shared__ float sRed[4];
  __shared__ float sba2;

  int t = threadIdx.x;
  if (t < 64) { sb1[t] = bias[t]; sb2[t] = bias[64 + t];
                sba1[t] = bias[128 + t]; swa2[t] = bias[192 + t]; }
  if (t == 0) sba2 = bias[384];
  __syncthreads();

  int wave = t >> 5, lane = t & 31;
  int lm = lane & 15, hi = lane >> 4;
  unsigned short* A = sA[wave];
  unsigned short* H = sH[wave];

  v16bf w2f[2][4];  // W2 fragments pinned in registers
  for (int kt = 0; kt < 2; ++kt)
    for (int nt = 0; nt < 4; ++nt)
      w2f[kt][nt] = load_frag_b(W2sw + (((kt * 4 + nt) * 32 + lane) << 4));

  float localMax = -3.402823466e38f;
  for (int tile = blockIdx.x * 4 + wave; tile < N_MACRO; tile += gridDim.x * 4) {
    int e0 = tile * 32;
    int sidx = senders[e0 + lane];
    int ridx = receivers[e0 + lane];
    // gather 32 rows of msg_in = [sender(64)|receiver(64)|edge(32)] -> bf16 LDS
    for (int r = 0; r < 32; ++r) {
      int sr = __shfl(sidx, r, 32);
      int rr = __shfl(ridx, r, 32);
      v2f sv = ((const v2f*)(nodeF + (long)sr * 64))[lane];
      v2f rv = ((const v2f*)(nodeF + (long)rr * 64))[lane];
      float ev = edgeF[(long)(e0 + r) * 32 + lane];
      *(unsigned*)(A + r * 168 + lane * 2) =
          (unsigned)f2bf(sv.x) | ((unsigned)f2bf(sv.y) << 16);
      *(unsigned*)(A + r * 168 + 64 + lane * 2) =
          (unsigned)f2bf(rv.x) | ((unsigned)f2bf(rv.y) << 16);
      A[r * 168 + 128 + lane] = f2bf(ev);
    }
    // message layer 1: [32x160]@[160x64], swish -> H
    for (int nt = 0; nt < 4; ++nt) {
      v8f c0 = {}, c1 = {};
      for (int kt = 0; kt < 5; ++kt) {
        v16bf b = load_frag_b(W1sw + (((kt * 4 + nt) * 32 + lane) << 4));
        v16bf a0 = load_frag_a(A + lm * 168 + kt * 32 + hi * 8);
        v16bf a1 = load_frag_a(A + (lm + 16) * 168 + kt * 32 + hi * 8);
        c0 = wmma_bf16(a0, b, c0);
        c1 = wmma_bf16(a1, b, c1);
      }
      float bb = sb1[nt * 16 + lm];
      for (int r = 0; r < 8; ++r) {
        H[(r + hi * 8) * 72 + nt * 16 + lm] = f2bf(swishf(c0[r] + bb));
        H[(16 + r + hi * 8) * 72 + nt * 16 + lm] = f2bf(swishf(c1[r] + bb));
      }
    }
    // message layer 2: [32x64]@[64x64] -> bf16 messages
    for (int nt = 0; nt < 4; ++nt) {
      v8f c0 = {}, c1 = {};
      for (int kt = 0; kt < 2; ++kt) {
        v16bf a0 = load_frag_a(H + lm * 72 + kt * 32 + hi * 8);
        v16bf a1 = load_frag_a(H + (lm + 16) * 72 + kt * 32 + hi * 8);
        c0 = wmma_bf16(a0, w2f[kt][nt], c0);
        c1 = wmma_bf16(a1, w2f[kt][nt], c1);
      }
      float bb = sb2[nt * 16 + lm];
      for (int r = 0; r < 8; ++r) {
        int m = r + hi * 8;
        msgs[(long)(e0 + m) * 64 + nt * 16 + lm] = f2bf(c0[r] + bb);
        msgs[(long)(e0 + 16 + m) * 64 + nt * 16 + lm] = f2bf(c1[r] + bb);
      }
    }
    // attention layer 1, swish -> H (reuse)
    for (int nt = 0; nt < 4; ++nt) {
      v8f c0 = {}, c1 = {};
      for (int kt = 0; kt < 5; ++kt) {
        v16bf b = load_frag_b(Wa1sw + (((kt * 4 + nt) * 32 + lane) << 4));
        v16bf a0 = load_frag_a(A + lm * 168 + kt * 32 + hi * 8);
        v16bf a1 = load_frag_a(A + (lm + 16) * 168 + kt * 32 + hi * 8);
        c0 = wmma_bf16(a0, b, c0);
        c1 = wmma_bf16(a1, b, c1);
      }
      float bb = sba1[nt * 16 + lm];
      for (int r = 0; r < 8; ++r) {
        H[(r + hi * 8) * 72 + nt * 16 + lm] = f2bf(swishf(c0[r] + bb));
        H[(16 + r + hi * 8) * 72 + nt * 16 + lm] = f2bf(swishf(c1[r] + bb));
      }
    }
    // logit = dot(H_row, Wa2) + ba2 ; one edge per lane, packed u32 reads
    float p = 0.0f;
    const unsigned short* hrow = H + lane * 72;
    for (int i = 0; i < 32; ++i) {
      unsigned hv = *(const unsigned*)(hrow + 2 * i);
      p += __uint_as_float(hv << 16) * swa2[2 * i] +
           __uint_as_float(hv & 0xFFFF0000u) * swa2[2 * i + 1];
    }
    float lg = p + sba2;
    logits[e0 + lane] = lg;
    float wm = lg;
    for (int o = 1; o < 32; o <<= 1) wm = fmaxf(wm, __shfl_xor(wm, o, 32));
    localMax = fmaxf(localMax, wm);
  }
  if (lane == 0) sRed[wave] = localMax;
  __syncthreads();
  if (t == 0) {  // one ordered-uint atomicMax per block
    float m = fmaxf(fmaxf(sRed[0], sRed[1]), fmaxf(sRed[2], sRed[3]));
    unsigned u = __float_as_uint(m);
    u = (u & 0x80000000u) ? ~u : (u | 0x80000000u);
    atomicMax(maxbits, u);
  }
}

// ------------------------------------------------------------ sum(exp) -----
// Also converts logits -> exp(l - max) in place so aggregation skips the exp.
__global__ __launch_bounds__(256) void sumexp_kernel(
    float* __restrict__ logits, const unsigned* __restrict__ scal,
    float* __restrict__ sumptr) {
  __shared__ float red[256];
  float mx = dec_max(scal[0]);
  float s = 0.0f;
  for (int i = blockIdx.x * 256 + threadIdx.x; i < N_EDGES; i += gridDim.x * 256) {
    float w = __expf(logits[i] - mx);
    logits[i] = w;
    s += w;
  }
  red[threadIdx.x] = s;
  __syncthreads();
  for (int o = 128; o > 0; o >>= 1) {
    if (threadIdx.x < o) red[threadIdx.x] += red[threadIdx.x + o];
    __syncthreads();
  }
  if (threadIdx.x == 0) atomicAdd(sumptr, red[0]);
}

// ------------------------------------------------- segment aggregation -----
// receivers sorted: 64 threads per node binary-search the edge range; no atomics.
__global__ __launch_bounds__(256) void aggregate_kernel(
    const int* __restrict__ receivers, const float* __restrict__ weights,
    const unsigned short* __restrict__ msgs, const unsigned* __restrict__ scal,
    float* __restrict__ agg) {
  int v = blockIdx.x * 4 + (threadIdx.x >> 6);
  int d = threadIdx.x & 63;
  float inv = __builtin_amdgcn_rcpf(((const float*)scal)[1]);
  int lo = 0, hic = N_EDGES;
  while (lo < hic) { int mid = (lo + hic) >> 1; if (receivers[mid] < v) lo = mid + 1; else hic = mid; }
  int s0 = lo;
  hic = N_EDGES;
  while (lo < hic) { int mid = (lo + hic) >> 1; if (receivers[mid] <= v) lo = mid + 1; else hic = mid; }
  int s1 = lo;
  float acc = 0.0f;
  for (int e = s0; e < s1; ++e)
    acc += weights[e] * bf2f(msgs[(long)e * 64 + d]);
  agg[(long)v * 64 + d] = acc * inv;
}

// ----------------------------------------------------- node update MLP -----
__global__ __launch_bounds__(256) void node_kernel(
    const float* __restrict__ nodeF, const float* __restrict__ agg,
    const unsigned short* __restrict__ Wu1sw,
    const unsigned short* __restrict__ Wu2sw, const float* __restrict__ bias,
    float* __restrict__ out) {
  __shared__ __align__(16) unsigned short sA[8][16 * 136];  // 16x128 bf16, padded
  __shared__ __align__(16) unsigned short sH[8][16 * 72];
  __shared__ float sbu1[64], sbu2[64];
  int t = threadIdx.x;
  if (t < 64) { sbu1[t] = bias[256 + t]; sbu2[t] = bias[320 + t]; }
  __syncthreads();
  int wave = t >> 5, lane = t & 31;
  int lm = lane & 15, hi = lane >> 4;
  unsigned short* A = sA[wave];
  unsigned short* H = sH[wave];
  v16bf w2f[2][4];
  for (int kt = 0; kt < 2; ++kt)
    for (int nt = 0; nt < 4; ++nt)
      w2f[kt][nt] = load_frag_b(Wu2sw + (((kt * 4 + nt) * 32 + lane) << 4));
  int tile = blockIdx.x * 8 + wave;
  if (tile >= NODE_TILES) return;
  int n0 = tile * 16;
  for (int r = 0; r < 16; ++r) {  // upd_in = [node(64)|aggregated(64)]
    v2f a = ((const v2f*)(nodeF + (long)(n0 + r) * 64))[lane];
    v2f g = ((const v2f*)(agg + (long)(n0 + r) * 64))[lane];
    *(unsigned*)(A + r * 136 + lane * 2) =
        (unsigned)f2bf(a.x) | ((unsigned)f2bf(a.y) << 16);
    *(unsigned*)(A + r * 136 + 64 + lane * 2) =
        (unsigned)f2bf(g.x) | ((unsigned)f2bf(g.y) << 16);
  }
  for (int nt = 0; nt < 4; ++nt) {
    v8f c = {};
    for (int kt = 0; kt < 4; ++kt) {
      v16bf a = load_frag_a(A + lm * 136 + kt * 32 + hi * 8);
      v16bf b = load_frag_b(Wu1sw + (((kt * 4 + nt) * 32 + lane) << 4));
      c = wmma_bf16(a, b, c);
    }
    float bb = sbu1[nt * 16 + lm];
    for (int r = 0; r < 8; ++r)
      H[(r + hi * 8) * 72 + nt * 16 + lm] = f2bf(swishf(c[r] + bb));
  }
  for (int nt = 0; nt < 4; ++nt) {
    v8f c = {};
    for (int kt = 0; kt < 2; ++kt) {
      v16bf a = load_frag_a(H + lm * 72 + kt * 32 + hi * 8);
      c = wmma_bf16(a, w2f[kt][nt], c);
    }
    float bb = sbu2[nt * 16 + lm];
    for (int r = 0; r < 8; ++r) {
      int m = r + hi * 8;
      out[(long)(n0 + m) * 64 + nt * 16 + lm] = c[r] + bb;
    }
  }
}

// ---------------------------------------------------------------------------
extern "C" void kernel_launch(void* const* d_in, const int* in_sizes, int n_in,
                              void* d_out, int out_size, void* d_ws, size_t ws_size,
                              hipStream_t stream) {
  const float* nodeF = (const float*)d_in[0];
  const float* edgeF = (const float*)d_in[1];
  const int* senders = (const int*)d_in[2];
  const int* receivers = (const int*)d_in[3];
  const float* W1 = (const float*)d_in[4];
  const float* b1 = (const float*)d_in[5];
  const float* W2 = (const float*)d_in[6];
  const float* b2 = (const float*)d_in[7];
  const float* Wa1 = (const float*)d_in[8];
  const float* ba1 = (const float*)d_in[9];
  const float* Wa2 = (const float*)d_in[10];
  const float* ba2 = (const float*)d_in[11];
  const float* Wu1 = (const float*)d_in[12];
  const float* bu1 = (const float*)d_in[13];
  const float* Wu2 = (const float*)d_in[14];
  const float* bu2 = (const float*)d_in[15];

  char* ws = (char*)d_ws;
  constexpr size_t OFF_BIAS   = 256;        // 385 f32: b1,b2,ba1,wa2,bu1,bu2,ba2
  constexpr size_t OFF_W1SW   = 2048;       // 20480 B
  constexpr size_t OFF_WA1SW  = 22528;      // 20480 B
  constexpr size_t OFF_W2SW   = 43008;      // 8192 B
  constexpr size_t OFF_WU1SW  = 51200;      // 16384 B
  constexpr size_t OFF_WU2SW  = 67584;      // 8192 B
  constexpr size_t OFF_LOGITS = 75776;      // E*4 = 6.4 MB
  constexpr size_t OFF_MSGS   = 6475776;    // E*64*2 = 204.8 MB
  constexpr size_t OFF_AGG    = 211275776;  // N*64*4 = 12.8 MB

  unsigned* scal = (unsigned*)ws;           // [0]=maxbits [1]=sum
  float* bias = (float*)(ws + OFF_BIAS);
  unsigned short* W1sw  = (unsigned short*)(ws + OFF_W1SW);
  unsigned short* Wa1sw = (unsigned short*)(ws + OFF_WA1SW);
  unsigned short* W2sw  = (unsigned short*)(ws + OFF_W2SW);
  unsigned short* Wu1sw = (unsigned short*)(ws + OFF_WU1SW);
  unsigned short* Wu2sw = (unsigned short*)(ws + OFF_WU2SW);
  float* logits = (float*)(ws + OFF_LOGITS);
  unsigned short* msgs = (unsigned short*)(ws + OFF_MSGS);
  float* agg = (float*)(ws + OFF_AGG);

  prep_kernel<<<64, 256, 0, stream>>>(W1, b1, W2, b2, Wa1, ba1, Wa2, ba2,
                                      Wu1, bu1, Wu2, bu2, scal, bias,
                                      W1sw, Wa1sw, W2sw, Wu1sw, Wu2sw);
  // 6250 blocks x 4 waves -> each wave exactly 2 macro-tiles
  edge_kernel<<<6250, 128, 0, stream>>>(nodeF, edgeF, senders, receivers,
                                        W1sw, Wa1sw, W2sw, bias, logits, msgs, scal);
  sumexp_kernel<<<1024, 256, 0, stream>>>(logits, scal, ((float*)scal) + 1);
  aggregate_kernel<<<N_NODES / 4, 256, 0, stream>>>(receivers, logits, msgs, scal, agg);
  node_kernel<<<(NODE_TILES + 7) / 8, 256, 0, stream>>>(nodeF, agg, Wu1sw, Wu2sw,
                                                        bias, (float*)d_out);
  (void)in_sizes; (void)n_in; (void)out_size; (void)ws_size;
}
